// Attention_69269232550541
// MI455X (gfx1250) — compile-verified, matrix operand order checked by reading, not run on hardware
//
#include <hip/hip_runtime.h>
#include <hip/hip_bf16.h>
#include <math.h>

#define B_ 32
#define S_ 4096
#define E_ 1024   // ENC2
#define D_ 512    // DEC
#define WROW (D_ + E_)   // W_attn row length = 1536
#define MB 128    // S-rows per block
#define NTHREADS 512     // 16 waves

typedef __attribute__((ext_vector_type(16))) __bf16 v16bf;
typedef __attribute__((ext_vector_type(8)))  __bf16 v8bf;
typedef __attribute__((ext_vector_type(8)))  float  v8f;
typedef __attribute__((ext_vector_type(4)))  float  v4f;

// ---------------------------------------------------------------------------
// Workspace layout (bytes):
//   [0, 1 MiB)                  : Wc in bf16, row-major (D_ x E_)
//   [1 MiB, 1 MiB + 64 KiB)     : h_proj f32 (B_ x D_)
//   [1 MiB + 64 KiB, +512 KiB)  : raw scores f32 (B_ x S_)
// ---------------------------------------------------------------------------

__device__ __forceinline__ float fast_tanh(float x) {
  // exact at +/-inf saturation: 1 - 2/(e^{2x}+1)
  float e = __expf(2.0f * x);
  return 1.0f - 2.0f * __builtin_amdgcn_rcpf(e + 1.0f);
}

// Build a 16-element bf16 WMMA fragment from two contiguous 16B chunks.
__device__ __forceinline__ v16bf load_frag(const __bf16* p) {
  v8bf lo = *(const v8bf*)p;
  v8bf hh = *(const v8bf*)(p + 16);
  v16bf r;
#pragma unroll
  for (int i = 0; i < 8; ++i) { r[i] = lo[i]; r[i + 8] = hh[i]; }
  return r;
}

// Convert Wc = W_attn[:, DEC:] to bf16 (stays hot in L2 for the main GEMM).
__global__ void wc_to_bf16_kernel(const float* __restrict__ W,
                                  __bf16* __restrict__ wc) {
  int idx = blockIdx.x * blockDim.x + threadIdx.x;   // 524288 total
  int d = idx >> 10;
  int e = idx & (E_ - 1);
  wc[idx] = (__bf16)W[(size_t)d * WROW + D_ + e];
}

// h_proj[b,d] = sum_k hidden[b,k] * W_attn[d,k] + b_attn[d]   (tiny GEMM, f32)
__global__ void hproj_kernel(const float* __restrict__ hidden,
                             const float* __restrict__ W,
                             const float* __restrict__ bias,
                             float* __restrict__ hp) {
  int idx = blockIdx.x * blockDim.x + threadIdx.x;   // 16384 total
  int b = idx >> 9;
  int d = idx & (D_ - 1);
  const float* hrow = hidden + (size_t)b * D_;
  const float* wrow = W + (size_t)d * WROW;
  float acc = bias[d];
#pragma unroll 8
  for (int k = 0; k < D_; k += 4) {
    v4f h = *(const v4f*)(hrow + k);
    v4f w = *(const v4f*)(wrow + k);
    acc += h[0]*w[0] + h[1]*w[1] + h[2]*w[2] + h[3]*w[3];
  }
  hp[idx] = acc;
}

// ---------------------------------------------------------------------------
// Main fused kernel: block = M 128 x N 512 (full DEC), 16 waves.
// K-loop unrolled x2 with ping-ponged LDS A buffers AND ping-ponged
// B-fragment register sets (no rotation copies). A-fragments pipelined
// one mt ahead. Fused tanh(.+h_proj) . v epilogue.
// ---------------------------------------------------------------------------
__launch_bounds__(NTHREADS, 1)
__global__ void attn_gemm_kernel(const float*  __restrict__ context,
                                 const __bf16* __restrict__ wc,
                                 const float*  __restrict__ hp,
                                 const float*  __restrict__ vvec,
                                 float*        __restrict__ scores) {
  __shared__ __attribute__((aligned(16))) __bf16 As[2][MB * 32]; // 2 x 8 KiB
  __shared__ float sScore[MB];

  const int tid  = threadIdx.x;
  const int lane = tid & 31;
  const int wave = tid >> 5;           // 0..15 -> N base = wave*32
  const int b    = blockIdx.y;
  const int s0   = blockIdx.x * MB;

  const int n  = lane & 15;            // column / row-in-tile index
  const int hi = lane >> 4;            // half-wave select
  const int kb = hi * 8;               // K sub-offset per 16-bit WMMA layout

  v8f acc[8][2] = {};                  // [m-tile][n-tile], 16x16 f32 each

  // Cooperative A staging coords: 128x32 f32 tile, 8 consecutive f32/thread.
  const int li = tid * 8;
  const int lr = li >> 5;              // row 0..127
  const int lc = li & 31;              // col, step 8
  const float* gctx =
      context + ((size_t)b * S_ + s0 + lr) * E_ + lc;

  // B (Wc^T) per-lane base: lane n -> output channel wave*32 + n.
  const __bf16* wcw = wc + (size_t)(wave * 32 + n) * E_ + kb;

  // Preload context tile 0 (non-temporal: one-shot 512MB stream, save L2).
  v4f f0 = __builtin_nontemporal_load((const v4f*)(gctx));
  v4f f1 = __builtin_nontemporal_load((const v4f*)(gctx + 4));

  // Preload B fragments for K-step 0.
  v16bf b00 = load_frag(wcw);
  v16bf b01 = load_frag(wcw + (size_t)16 * E_);
  v16bf b10, b11;

#define STAGE_STEP(BUF)                                                     \
  {                                                                         \
    v8bf h8;                                                                \
    h8[0] = (__bf16)f0[0]; h8[1] = (__bf16)f0[1];                           \
    h8[2] = (__bf16)f0[2]; h8[3] = (__bf16)f0[3];                           \
    h8[4] = (__bf16)f1[0]; h8[5] = (__bf16)f1[1];                           \
    h8[6] = (__bf16)f1[2]; h8[7] = (__bf16)f1[3];                           \
    *(v8bf*)(&As[BUF][lr * 32 + lc]) = h8;                                  \
  }                                                                         \
  __syncthreads();

#define COMPUTE_STEP(BUF, BF0, BF1)                                         \
  {                                                                         \
    v16bf a0 = load_frag(&As[BUF][n * 32 + kb]);                            \
    _Pragma("unroll")                                                       \
    for (int mt = 0; mt < 8; ++mt) {                                        \
      v16bf a1;                                                             \
      if (mt < 7) a1 = load_frag(&As[BUF][((mt + 1) * 16 + n) * 32 + kb]);  \
      acc[mt][0] = __builtin_amdgcn_wmma_f32_16x16x32_bf16(                 \
          false, a0, false, BF0, (short)0, acc[mt][0], false, false);       \
      acc[mt][1] = __builtin_amdgcn_wmma_f32_16x16x32_bf16(                 \
          false, a0, false, BF1, (short)0, acc[mt][1], false, false);       \
      if (mt < 7) a0 = a1;                                                  \
    }                                                                       \
  }

  for (int k = 0; k < E_; k += 64) {
    // ---- half-step A: LDS buf 0, consumes b00/b01, prefetches k+32 ----
    STAGE_STEP(0)
    f0 = __builtin_nontemporal_load((const v4f*)(gctx + k + 32));
    f1 = __builtin_nontemporal_load((const v4f*)(gctx + k + 36));
    b10 = load_frag(wcw + k + 32);
    b11 = load_frag(wcw + (size_t)16 * E_ + k + 32);
    COMPUTE_STEP(0, b00, b01)

    // ---- half-step B: LDS buf 1, consumes b10/b11, prefetches k+64 ----
    STAGE_STEP(1)
    if (k + 64 < E_) {
      f0 = __builtin_nontemporal_load((const v4f*)(gctx + k + 64));
      f1 = __builtin_nontemporal_load((const v4f*)(gctx + k + 68));
      b00 = load_frag(wcw + k + 64);
      b01 = load_frag(wcw + (size_t)16 * E_ + k + 64);
    }
    COMPUTE_STEP(1, b10, b11)
  }
#undef STAGE_STEP
#undef COMPUTE_STEP

  // --- fused epilogue: scores[row] = sum_d tanh(c_proj + h_proj[d]) * v[d] ---
  if (tid < MB) sScore[tid] = 0.0f;
  __syncthreads();

  float hpv[2], vv[2];
#pragma unroll
  for (int nt = 0; nt < 2; ++nt) {
    int ng = wave * 32 + nt * 16 + n;
    hpv[nt] = hp[(size_t)b * D_ + ng];
    vv[nt]  = vvec[ng];
  }

#pragma unroll
  for (int mt = 0; mt < 8; ++mt) {
#pragma unroll
    for (int r = 0; r < 8; ++r) {
      float s = fast_tanh(acc[mt][0][r] + hpv[0]) * vv[0]
              + fast_tanh(acc[mt][1][r] + hpv[1]) * vv[1];
      // reduce over the 16 columns held by this half-wave
      s += __shfl_xor(s, 1, 32);
      s += __shfl_xor(s, 2, 32);
      s += __shfl_xor(s, 4, 32);
      s += __shfl_xor(s, 8, 32);
      if (n == 0)
        atomicAdd(&sScore[mt * 16 + r + hi * 8], s);  // ds_add_f32 across waves
    }
  }
  __syncthreads();

  if (tid < MB)
    scores[(size_t)b * S_ + s0 + tid] = sScore[tid];
}

// Masked softmax over S per batch row.
__global__ void softmax_kernel(const float* __restrict__ scores,
                               const int*   __restrict__ mask,
                               float*       __restrict__ out) {
  __shared__ float red[256];
  const int b = blockIdx.x, tid = threadIdx.x;
  const float* sr = scores + (size_t)b * S_;
  const int*   mr = mask   + (size_t)b * S_;

  float mx = -INFINITY;
  for (int i = tid; i < S_; i += 256) {
    float v = (mr[i] == 0) ? -INFINITY : sr[i];
    mx = fmaxf(mx, v);
  }
  red[tid] = mx; __syncthreads();
  for (int s = 128; s > 0; s >>= 1) {
    if (tid < s) red[tid] = fmaxf(red[tid], red[tid + s]);
    __syncthreads();
  }
  mx = red[0]; __syncthreads();

  float sum = 0.0f;
  for (int i = tid; i < S_; i += 256)
    sum += (mr[i] == 0) ? 0.0f : __expf(sr[i] - mx);
  red[tid] = sum; __syncthreads();
  for (int s = 128; s > 0; s >>= 1) {
    if (tid < s) red[tid] += red[tid + s];
    __syncthreads();
  }
  sum = red[0];

  const float inv = 1.0f / sum;
  for (int i = tid; i < S_; i += 256) {
    float v = (mr[i] == 0) ? 0.0f : __expf(sr[i] - mx);
    out[(size_t)b * S_ + i] = v * inv;
  }
}

extern "C" void kernel_launch(void* const* d_in, const int* in_sizes, int n_in,
                              void* d_out, int out_size, void* d_ws, size_t ws_size,
                              hipStream_t stream) {
  (void)in_sizes; (void)n_in; (void)out_size; (void)ws_size;

  const float* hidden  = (const float*)d_in[0];
  const float* context = (const float*)d_in[1];
  const int*   mask    = (const int*)  d_in[2];
  const float* W       = (const float*)d_in[3];
  const float* bias    = (const float*)d_in[4];
  const float* v       = (const float*)d_in[5];
  float* out = (float*)d_out;

  char*   ws     = (char*)d_ws;
  __bf16* wcBf   = (__bf16*)ws;                                    // 1 MiB
  float*  hp     = (float*)(ws + (size_t)D_ * E_ * 2);             // 64 KiB
  float*  scores = (float*)(ws + (size_t)D_ * E_ * 2
                               + (size_t)B_ * D_ * 4);             // 512 KiB

  wc_to_bf16_kernel<<<(D_ * E_) / 256, 256, 0, stream>>>(W, wcBf);
  hproj_kernel<<<(B_ * D_) / 256, 256, 0, stream>>>(hidden, W, bias, hp);

  dim3 grid(S_ / MB, B_);
  attn_gemm_kernel<<<grid, NTHREADS, 0, stream>>>(context, wcBf, hp, v, scores);

  softmax_kernel<<<B_, 256, 0, stream>>>(scores, mask, out);
}
